// TurboQuantizer_79826262163490
// MI455X (gfx1250) — matrix operand order repaired and checked by reading, not compiled, not run on hardware
//
#include <hip/hip_runtime.h>
#include <math.h>

#define DIM 1024
#define NROWS 32768
#define COEFF (1.25331413731550025f / 1024.0f)   // sqrt(pi/2)/DIM

typedef __attribute__((ext_vector_type(16))) __bf16        v16bf;
typedef __attribute__((ext_vector_type(8)))  float         v8f;
typedef __attribute__((ext_vector_type(4)))  unsigned int  u32x4;
typedef __attribute__((ext_vector_type(4)))  int           i32x4;
typedef __attribute__((ext_vector_type(8)))  int           i32x8;

union Frag32 {            // 32 bytes: one WMMA bf16 operand fragment
  u32x4 u[2];
  v16bf v;
};

__device__ __forceinline__ __bf16 f2bf(float f) {
  unsigned u = __builtin_bit_cast(unsigned, f);
  unsigned r = u + 0x7FFFu + ((u >> 16) & 1u);     // round-to-nearest-even
  unsigned short h = (unsigned short)(r >> 16);
  return __builtin_bit_cast(__bf16, h);
}

__device__ __forceinline__ float f16rt(float f) {  // fp16 store/decode roundtrip
  return (float)(_Float16)f;
}

// ---------------------------------------------------------------------------
// Kernel 0: proj f32 [D,D] -> bf16 row-major (projB) and bf16 transposed (projT)
// ---------------------------------------------------------------------------
__global__ void proj_convert_kernel(const float* __restrict__ proj,
                                    __bf16* __restrict__ projB,
                                    __bf16* __restrict__ projT) {
  __shared__ float tile[32][33];
  const int tx = threadIdx.x, ty = threadIdx.y;         // 32 x 8
  const int e0 = blockIdx.y * 32, d0 = blockIdx.x * 32;
#pragma unroll
  for (int i = 0; i < 32; i += 8) {
    int e = e0 + ty + i, d = d0 + tx;
    float v = proj[(size_t)e * DIM + d];
    tile[ty + i][tx] = v;
    projB[(size_t)e * DIM + d] = f2bf(v);
  }
  __syncthreads();
#pragma unroll
  for (int i = 0; i < 32; i += 8) {
    int d = d0 + ty + i, e = e0 + tx;
    projT[(size_t)d * DIM + e] = f2bf(tile[tx][ty + i]);
  }
}

// ---------------------------------------------------------------------------
// In-LDS fast Walsh-Hadamard transform, 1024 points, 256 threads
// ---------------------------------------------------------------------------
__device__ __forceinline__ void fwht1024(float* s, int t) {
  for (int len = 1; len < DIM; len <<= 1) {
    __syncthreads();
#pragma unroll
    for (int rep = 0; rep < 2; ++rep) {
      int p = t + rep * 256;                       // 512 disjoint butterflies
      int i = ((p & ~(len - 1)) << 1) | (p & (len - 1));
      float a = s[i], b = s[i + len];
      s[i]       = a + b;
      s[i + len] = a - b;
    }
  }
  __syncthreads();
}

// ---------------------------------------------------------------------------
// Kernel 1: per-row normalize + RHT + 3-bit quantize + inverse RHT + residual
// ---------------------------------------------------------------------------
__global__ void __launch_bounds__(256)
fwht_quant_kernel(const float* __restrict__ x,
                  const float* __restrict__ dsg,
                  const float* __restrict__ codebook,    // 8
                  const float* __restrict__ boundaries,  // 7
                  __bf16* __restrict__ residual,
                  float*  __restrict__ mse,
                  float*  __restrict__ norms_q,
                  float*  __restrict__ rnorms_q) {
  __shared__ float s[DIM];
  __shared__ float red[256];
  const int t = threadIdx.x;
  const size_t n = blockIdx.x;
  const float scale = 0.03125f;                    // 1/sqrt(1024)

  float xv[4], ds4[4], un[4];
  float ssq = 0.f;
#pragma unroll
  for (int j = 0; j < 4; ++j) {
    int p = t + j * 256;
    float v = x[n * DIM + p];
    xv[j] = v;
    ds4[j] = dsg[p];
    ssq += v * v;
  }
  red[t] = ssq; __syncthreads();
  for (int off = 128; off > 0; off >>= 1) {
    if (t < off) red[t] += red[t + off];
    __syncthreads();
  }
  float norm = sqrtf(red[0]);
  __syncthreads();
  float invn = (norm > 0.f) ? (1.f / fmaxf(norm, 1e-12f)) : 0.f;

#pragma unroll
  for (int j = 0; j < 4; ++j) {
    un[j] = xv[j] * invn;
    s[t + j * 256] = un[j] * ds4[j];
  }

  fwht1024(s, t);                                  // rotated (pre-scale)

  float bnd[7], cb[8];
#pragma unroll
  for (int k = 0; k < 7; ++k) bnd[k] = boundaries[k];
#pragma unroll
  for (int k = 0; k < 8; ++k) cb[k] = codebook[k];

  float qv[4];
#pragma unroll
  for (int j = 0; j < 4; ++j) {
    float v = s[t + j * 256] * scale;
    // searchsorted(left): idx = #{bnd < v}; monotone codebook -> select chain
    float q = cb[0];
#pragma unroll
    for (int k = 0; k < 7; ++k) q = (bnd[k] < v) ? cb[k + 1] : q;
    qv[j] = q;
  }
  __syncthreads();
#pragma unroll
  for (int j = 0; j < 4; ++j) s[t + j * 256] = qv[j];

  fwht1024(s, t);                                  // inverse rotation

  float rsq = 0.f;
#pragma unroll
  for (int j = 0; j < 4; ++j) {
    int p = t + j * 256;
    float m = s[p] * scale * ds4[j];               // mse_unit
    float r = un[j] - m;                           // residual
    rsq += r * r;
    mse[n * DIM + p]      = m;
    residual[n * DIM + p] = f2bf(r);
  }
  red[t] = rsq; __syncthreads();
  for (int off = 128; off > 0; off >>= 1) {
    if (t < off) red[t] += red[t + off];
    __syncthreads();
  }
  if (t == 0) {
    norms_q[n]  = f16rt(norm);
    rnorms_q[n] = f16rt(sqrtf(red[0]));
  }
}

// ---------------------------------------------------------------------------
// Stage one 16 x 1024 bf16 A-panel (32 KB, contiguous) into LDS.
// Preferred path: Tensor Data Mover (1D descriptor, TENSORcnt).
// Fallback: cooperative b128 copy.
// ---------------------------------------------------------------------------
__device__ __forceinline__ void stage_A_panel(const __bf16* __restrict__ gsrc,
                                              __bf16* __restrict__ lA, int t) {
#if __has_builtin(__builtin_amdgcn_tensor_load_to_lds)
  if (t < 32) {                                    // wave 0 issues the TDM op
    unsigned lds_addr = (unsigned)(size_t)lA;      // flat addr low 32 = LDS offset
    unsigned long long ga = (unsigned long long)(size_t)gsrc;
    // D# group 0: count=1 | lds_addr | global_addr[56:0] | type=2
    u32x4 g0 = { 0x1u,
                 lds_addr,
                 (unsigned)(ga & 0xFFFFFFFFull),
                 (unsigned)((ga >> 32) & 0x01FFFFFFull) | 0x80000000u };
    // D# group 1: wg_mask=0, data_size=2B(code 1), tensor_dim0=16384,
    // tensor_dim1=1, tile_dim0=16384, tile_dim1=1, dim0_stride=16384
    i32x8 g1 = { (int)0x00010000,   // data_size=1 << 16
                 (int)0x40000000,   // tensor_dim0[15:0]=0x4000 << 16
                 (int)0x00010000,   // tensor_dim1[15:0]=1 << 16
                 (int)0x40000000,   // tile_dim0=0x4000 << 16
                 (int)0x00000001,   // tile_dim1=1
                 (int)16384,        // tensor_dim0_stride[31:0]
                 0, 0 };
    i32x4 g2 = {};                  // <=2D tensor: groups 2/3 zero
    i32x4 g3 = {};
#if defined(__clang_major__) && (__clang_major__ >= 23)
    i32x8 g4 = {};
    __builtin_amdgcn_tensor_load_to_lds(g0, g1, g2, g3, g4, 0);
#else
    __builtin_amdgcn_tensor_load_to_lds(g0, g1, g2, g3, 0);
#endif
#if __has_builtin(__builtin_amdgcn_s_wait_tensorcnt)
    __builtin_amdgcn_s_wait_tensorcnt(0);
#else
    asm volatile("s_wait_tensorcnt 0x0" ::: "memory");
#endif
  }
#else
  const u32x4* src = (const u32x4*)gsrc;
  u32x4* dst = (u32x4*)lA;
#pragma unroll
  for (int i = 0; i < 8; ++i) dst[t + i * 256] = src[t + i * 256];
#endif
}

// ---------------------------------------------------------------------------
// Kernels 2/3: bf16 WMMA GEMM, C[m, col] = sum_k A[m,k] * Bm[col,k]
//   block = 8 waves; block tile = 16 (M) x 1024 (all N): each wave owns 8
//   16-wide column subtiles (8 x v8f accumulators), sharing one LDS A-panel.
//   A is read from HBM exactly once per GEMM; B (2 MB) stays L2-resident.
//   mode 0: epilogue = sign -> +-1 bf16       (sr = residual @ proj^T)
//   mode 1: epilogue = (mse + c*rn*acc)*nm    (dir = signs @ proj, fused out)
// Fragment packing: A 16x32 / B 32x16 in contiguous K (k = 16*(lane/16) + s), C 16x16 per ISA.
// ---------------------------------------------------------------------------
__global__ void __launch_bounds__(256)
qjl_gemm_kernel(const __bf16* __restrict__ A,      // [NROWS, DIM]
                const __bf16* __restrict__ Bm,     // [DIM, DIM], row=out col, contiguous K
                __bf16* __restrict__ signs_out,    // mode 0
                const float* __restrict__ mse,     // mode 1
                const float* __restrict__ norms,   // mode 1
                const float* __restrict__ rnorms,  // mode 1
                float* __restrict__ out,           // mode 1
                int mode) {
  __shared__ __align__(16) __bf16 lA[16 * DIM];    // 32 KB A K-panel

  const int t    = threadIdx.x;
  const int wave = t >> 5, lane = t & 31;
  const int m0   = blockIdx.x * 16;

  stage_A_panel(A + (size_t)m0 * DIM, lA, t);
  __syncthreads();

  const int arow  = lane & 15;
  const int ahalf = (lane < 16) ? 0 : 16;          // A lane K sub-offset
  const int bhalf = (lane < 16) ? 0 : 16;          // B lane K sub-offset
  // wave's first column subtile; subtile j adds 128 columns
  const int col0  = wave * 16 + (lane & 15);
  const __bf16* bbase = Bm + (size_t)col0 * DIM + bhalf;

  v8f acc[8];
#pragma unroll
  for (int j = 0; j < 8; ++j) acc[j] = {};

  for (int kb = 0; kb < DIM / 32; ++kb) {
    const int k0 = kb * 32;
    Frag32 af;
    const u32x4* ap = (const u32x4*)(lA + arow * DIM + k0 + ahalf);
    af.u[0] = ap[0];                               // K k0+{0..7}(+ahalf)
    af.u[1] = ap[1];                               // K k0+8+{0..7}(+ahalf)
#pragma unroll
    for (int j = 0; j < 8; ++j) {
      Frag32 bfr;
      const u32x4* bp = (const u32x4*)(bbase + (size_t)j * 128 * DIM + k0);
      bfr.u[0] = bp[0];                            // 16 contiguous K halves
      bfr.u[1] = bp[1];
      acc[j] = __builtin_amdgcn_wmma_f32_16x16x32_bf16(false, af.v, false, bfr.v,
                                                       (short)0, acc[j], false, false);
    }
  }

  if (mode == 0) {
#pragma unroll
    for (int r = 0; r < 8; ++r) {
      int row = m0 + r + ((lane >> 4) << 3);
#pragma unroll
      for (int j = 0; j < 8; ++j) {
        int col = col0 + j * 128;
        unsigned short bits = (acc[j][r] >= 0.f) ? (unsigned short)0x3F80u
                                                 : (unsigned short)0xBF80u;
        signs_out[(size_t)row * DIM + col] = __builtin_bit_cast(__bf16, bits);
      }
    }
  } else {
#pragma unroll
    for (int r = 0; r < 8; ++r) {
      int row = m0 + r + ((lane >> 4) << 3);
      float nm = norms[row];
      float cq = COEFF * rnorms[row];
#pragma unroll
      for (int j = 0; j < 8; ++j) {
        size_t idx = (size_t)row * DIM + col0 + j * 128;
        out[idx] = (mse[idx] + cq * acc[j][r]) * nm;
      }
    }
  }
}

// ---------------------------------------------------------------------------
extern "C" void kernel_launch(void* const* d_in, const int* in_sizes, int n_in,
                              void* d_out, int out_size, void* d_ws, size_t ws_size,
                              hipStream_t stream) {
  (void)in_sizes; (void)n_in; (void)out_size; (void)ws_size;
  const float* x        = (const float*)d_in[0];
  // d_in[1] = hadamard matrix: unused, replaced by in-LDS FWHT
  const float* d_signs  = (const float*)d_in[2];
  const float* codebook = (const float*)d_in[3];
  const float* bnds     = (const float*)d_in[4];
  const float* proj     = (const float*)d_in[5];
  float* out = (float*)d_out;

  char* ws = (char*)d_ws;
  const size_t ND = (size_t)NROWS * DIM;           // 33,554,432
  __bf16* residual = (__bf16*)(ws);                              // 64 MiB
  float*  mse      = (float*)(ws + ND * 2);                      // 128 MiB
  __bf16* signs    = (__bf16*)(ws + ND * 6);                     // 64 MiB
  __bf16* projB    = (__bf16*)(ws + ND * 8);                     // 2 MiB
  __bf16* projT    = (__bf16*)(ws + ND * 8 + (size_t)DIM * DIM * 2);
  float*  norms    = (float*)(ws + ND * 8 + (size_t)DIM * DIM * 4);
  float*  rnorms   = (float*)(ws + ND * 8 + (size_t)DIM * DIM * 4 + (size_t)NROWS * 4);

  proj_convert_kernel<<<dim3(DIM / 32, DIM / 32), dim3(32, 8), 0, stream>>>(
      proj, projB, projT);

  fwht_quant_kernel<<<NROWS, 256, 0, stream>>>(
      x, d_signs, codebook, bnds, residual, mse, norms, rnorms);

  // GEMM 1: sr = residual @ proj^T  -> signs (+-1, bf16)
  qjl_gemm_kernel<<<dim3(NROWS / 16), 256, 0, stream>>>(
      residual, projB, signs, nullptr, nullptr, nullptr, nullptr, 0);

  // GEMM 2: direction = signs @ proj, fused epilogue -> out
  qjl_gemm_kernel<<<dim3(NROWS / 16), 256, 0, stream>>>(
      signs, projT, nullptr, mse, norms, rnorms, out, 1);
}
